// HierarchicalGPT_48713519071309
// MI455X (gfx1250) — matrix operand order repaired and checked by reading, not compiled, not run on hardware
//
#include <hip/hip_runtime.h>
#include <hip/hip_bf16.h>
#include <math.h>

// ---------------------------------------------------------------------------
// HierarchicalGPT forward for MI455X (gfx1250, wave32, WMMA).
// All GEMM operands are pre-converted to f16 (one pass) so the WMMA tile loop
// is pure data movement: 16B global loads -> 16B LDS stores -> ds_load_b128
// fragments -> v_wmma_f32_16x16x32_f16, software-pipelined one K-tile ahead.
// ---------------------------------------------------------------------------

typedef __attribute__((ext_vector_type(16))) _Float16 v16h;
typedef __attribute__((ext_vector_type(8)))  _Float16 h8;
typedef __attribute__((ext_vector_type(4)))  _Float16 h4;
typedef __attribute__((ext_vector_type(8)))  float    v8f;
typedef __attribute__((ext_vector_type(4)))  float    f4;

#define BB   2
#define TT   2048
#define VV   50257
#define DWD  256
#define DMD  384
#define LBSD 64

#define BM 64
#define BN 64
#define BK 32

#define FLAG_NT   1
#define FLAG_HALF 2

// ======================= f32 -> f16 bulk convert (vectorized) ==============
__global__ void k_tof16(const float* __restrict__ X, _Float16* __restrict__ Y, int n4)
{
    int i = blockIdx.x * blockDim.x + threadIdx.x;
    if (i >= n4) return;
    f4 v = ((const f4*)X)[i];
    h4 h;
#pragma unroll
    for (int e = 0; e < 4; ++e) h[e] = (_Float16)v[e];
    ((h4*)Y)[i] = h;
}

// ======================= WMMA tiled GEMM: Y = act(A @ W^T + b) (+resid) ====
// A: [M,K] f16 row-major; W: [N,K] f16 row-major. M%64==0, K%32==0, N free.
// 256 threads = 8 waves; wave (wm,wn) computes a 16x32 sub-tile of a 64x64
// block tile. Staging is register double-buffered: loads for tile k+1 overlap
// the ds_load/WMMA of tile k; tile k+2 is prefetched into L2.
__global__ __launch_bounds__(256)
void k_gemm(const _Float16* __restrict__ A, const _Float16* __restrict__ W,
            const float* __restrict__ bias, const float* __restrict__ resid,
            void* __restrict__ Y, int M, int N, int K, int act, int flags)
{
    __shared__ _Float16 As[BM][BK];
    __shared__ _Float16 Bs[BN][BK];

    const int tid  = threadIdx.x;
    const int lane = tid & 31;
    const int wave = tid >> 5;
    const int wm   = wave & 3;          // 4 M-subtiles of 16
    const int wn   = wave >> 2;         // 2 N-subtiles of 32
    const int half = lane >> 4;
    const int l16  = lane & 15;
    const int m0   = blockIdx.y * BM;
    const int n0   = blockIdx.x * BN;

    // Each thread stages one 8-half (16B) segment of the 64x32 tiles.
    const int sr = tid >> 2;            // 0..63 : tile row
    const int sc = (tid & 3) << 3;      // 0,8,16,24 : k offset (halves)
    const _Float16* Arow = A + (size_t)(m0 + sr) * K + sc;
    const int gn_s = n0 + sr;
    const _Float16* Wrow = W + (size_t)(gn_s < N ? gn_s : 0) * K + sc;

    const h8 hz = {};
    h8 areg = *(const h8*)Arow;
    h8 breg = (gn_s < N) ? *(const h8*)Wrow : hz;

    v8f c0 = {}, c1 = {};

    for (int kb = 0; kb < K; kb += BK) {
        *(h8*)&As[sr][sc] = areg;
        *(h8*)&Bs[sr][sc] = breg;
        __syncthreads();

        // Pipeline: issue next tile's global loads before consuming LDS.
        const int kn = kb + BK;
        if (kn < K) {
            areg = *(const h8*)(Arow + kn);
            breg = (gn_s < N) ? *(const h8*)(Wrow + kn) : hz;
        }
        if (kb + 2 * BK < K) {
            __builtin_prefetch(Arow + kb + 2 * BK, 0, 1);
            __builtin_prefetch(Wrow + kb + 2 * BK, 0, 1);
        }

        // Fragments in contiguous k order k = 16*half + e, same for A and B (A 16x32 ; B 32x16).
        v16h af, bf0, bf1;
#pragma unroll
        for (int e = 0; e < 16; ++e) {
            int ka = (half << 4) + e;
            af[e]  = As[wm * 16 + l16][ka];
            int kbb = (half << 4) + e;
            bf0[e] = Bs[wn * 32 + l16][kbb];
            bf1[e] = Bs[wn * 32 + 16 + l16][kbb];
        }
        c0 = __builtin_amdgcn_wmma_f32_16x16x32_f16(false, af, false, bf0, (short)0, c0, false, false);
        c1 = __builtin_amdgcn_wmma_f32_16x16x32_f16(false, af, false, bf1, (short)0, c1, false, false);
        __syncthreads();
    }

    // Epilogue: lane holds col n=l16, rows m = r + 8*half.
    const bool nt = flags & FLAG_NT, ho = flags & FLAG_HALF;
#pragma unroll
    for (int r = 0; r < 8; ++r) {
        int gm = m0 + wm * 16 + half * 8 + r;
#pragma unroll
        for (int nn = 0; nn < 2; ++nn) {
            int gn = n0 + wn * 32 + nn * 16 + l16;
            if (gn < N) {
                size_t o = (size_t)gm * N + gn;
                float v = nn ? c1[r] : c0[r];
                if (bias)  v += bias[gn];
                if (act == 1) v = 0.5f * v * (1.0f + erff(v * 0.70710678118654752f)); // exact GELU
                if (resid) v += resid[o];
                if (ho)      ((_Float16*)Y)[o] = (_Float16)v;
                else if (nt) __builtin_nontemporal_store(v, (float*)Y + o);
                else         ((float*)Y)[o] = v;
            }
        }
    }
}

// ======================= Flash attention (WMMA, one wave / 16 q-rows) ======
// Q,K,V f16 laid out [B,H,T,hd]; Y f32 [B,T,H*hd]. hd in {48,64} (padded).
__global__ __launch_bounds__(32)
void k_attn(const _Float16* __restrict__ Q, const _Float16* __restrict__ Km,
            const _Float16* __restrict__ V, float* __restrict__ Y,
            int Bv, int H, int Tv, int hd)
{
    __shared__ _Float16 p16[16][32];

    const int lane = threadIdx.x;
    const int half = lane >> 4, l16 = lane & 15;
    const int q0 = blockIdx.x * 16;
    const int h  = blockIdx.y;
    const int b  = blockIdx.z;
    const float scale = rsqrtf((float)hd);
    const size_t bh = ((size_t)b * H + h) * (size_t)Tv * hd;
    const h8 hz = {};

    // Q fragments: contiguous 8-half runs (16B loads), zero-padded to 64.
    const _Float16* qrow = Q + bh + (size_t)(q0 + l16) * hd;
    v16h qf[2];
#pragma unroll
    for (int f = 0; f < 2; ++f)
#pragma unroll
        for (int c = 0; c < 2; ++c) {
            int kbase = f * 32 + half * 16 + c * 8;
            h8 qc = (kbase < hd) ? *(const h8*)(qrow + kbase) : hz;
#pragma unroll
            for (int j = 0; j < 8; ++j) qf[f][c * 8 + j] = qc[j];
        }

    v8f o[4] = {};                 // 4 hd-subtiles of 16
    float rmax[8], rsum[8];
#pragma unroll
    for (int r = 0; r < 8; ++r) { rmax[r] = -1e30f; rsum[r] = 0.0f; }

    for (int k0 = 0; k0 <= q0 + 15; k0 += 32) {
        // S = q @ k^T for 32 keys (two 16-key C tiles); K rows are 16-half runs.
        v8f s[2] = {};
#pragma unroll
        for (int sub = 0; sub < 2; ++sub) {
            const _Float16* krow = Km + bh + (size_t)(k0 + sub * 16 + l16) * hd;
#pragma unroll
            for (int f = 0; f < 2; ++f) {
                int kbase = f * 32 + half * 16;
                h8 k0h = hz, k1h = hz;
                if (kbase < hd) { k0h = *(const h8*)(krow + kbase); k1h = *(const h8*)(krow + kbase + 8); }
                v16h bf;
#pragma unroll
                for (int j = 0; j < 8; ++j) { bf[j] = k0h[j]; bf[8 + j] = k1h[j]; }
                s[sub] = __builtin_amdgcn_wmma_f32_16x16x32_f16(false, qf[f], false, bf, (short)0, s[sub], false, false);
            }
        }
        // Causal mask + online softmax (1/sqrt(hd) applied here, in f32).
#pragma unroll
        for (int r = 0; r < 8; ++r) {
            int m = q0 + half * 8 + r;
            float s0v = (k0 + l16      <= m) ? s[0][r] * scale : -1e30f;
            float s1v = (k0 + 16 + l16 <= m) ? s[1][r] * scale : -1e30f;
            float mx = fmaxf(s0v, s1v);
            for (int d = 8; d >= 1; d >>= 1) mx = fmaxf(mx, __shfl_xor(mx, d, 32));
            float nm   = fmaxf(rmax[r], mx);
            float corr = expf(rmax[r] - nm);
            float p0 = expf(s0v - nm), p1 = expf(s1v - nm);
            float ps = p0 + p1;
            for (int d = 8; d >= 1; d >>= 1) ps += __shfl_xor(ps, d, 32);
            rsum[r] = rsum[r] * corr + ps;
            rmax[r] = nm;
#pragma unroll
            for (int nt = 0; nt < 4; ++nt) o[nt][r] *= corr;
            p16[half * 8 + r][l16]      = (_Float16)p0;
            p16[half * 8 + r][16 + l16] = (_Float16)p1;
        }
        __syncthreads();   // single-wave WG: lowers to NOP, orders LDS

        // P as A-fragment; V as B-fragments (columnar 16-bit gather).
        v16h pa;
#pragma unroll
        for (int e = 0; e < 16; ++e)
            pa[e] = p16[l16][(half << 4) + e];
#pragma unroll
        for (int nt = 0; nt < 4; ++nt) {
            int col = nt * 16 + l16;
            v16h vb;
#pragma unroll
            for (int e = 0; e < 16; ++e)
                vb[e] = (col < hd) ? V[bh + (size_t)(k0 + (half << 4) + e) * hd + col] : (_Float16)0.0f;
            o[nt] = __builtin_amdgcn_wmma_f32_16x16x32_f16(false, pa, false, vb, (short)0, o[nt], false, false);
        }
        __syncthreads();
    }

#pragma unroll
    for (int nt = 0; nt < 4; ++nt) {
        int col = nt * 16 + l16;
        if (col < hd)
#pragma unroll
            for (int r = 0; r < 8; ++r) {
                int t = q0 + half * 8 + r;
                Y[((size_t)b * Tv + t) * (size_t)(H * hd) + h * hd + col] = o[nt][r] / rsum[r];
            }
    }
}

// ======================= LayerNorm (one block per row), f16 or f32 out =====
__global__ __launch_bounds__(256)
void k_ln(const float* __restrict__ X, const float* __restrict__ g,
          const float* __restrict__ bb, void* __restrict__ Yp, int D, int out_half)
{
    __shared__ float red[256];
    const int row = blockIdx.x;
    const float* x = X + (size_t)row * D;

    float s = 0.0f;
    for (int j = threadIdx.x; j < D; j += 256) s += x[j];
    red[threadIdx.x] = s; __syncthreads();
    for (int o = 128; o > 0; o >>= 1) { if (threadIdx.x < o) red[threadIdx.x] += red[threadIdx.x + o]; __syncthreads(); }
    float mean = red[0] / D; __syncthreads();

    float s2 = 0.0f;
    for (int j = threadIdx.x; j < D; j += 256) { float d = x[j] - mean; s2 += d * d; }
    red[threadIdx.x] = s2; __syncthreads();
    for (int o = 128; o > 0; o >>= 1) { if (threadIdx.x < o) red[threadIdx.x] += red[threadIdx.x + o]; __syncthreads(); }
    float inv = rsqrtf(red[0] / D + 1e-5f);

    for (int j = threadIdx.x; j < D; j += 256) {
        float r = (x[j] - mean) * inv * g[j] + bb[j];
        if (out_half) ((_Float16*)Yp)[(size_t)row * D + j] = (_Float16)r;
        else          ((float*)Yp)[(size_t)row * D + j] = r;
    }
}

// ======================= RoPE + qkv split to [B,H,T,hd] (f16 in/out) =======
__global__ void k_rope_split(const _Float16* __restrict__ QKV,
                             _Float16* __restrict__ Qo, _Float16* __restrict__ Ko,
                             _Float16* __restrict__ Vo, int Bv, int H, int Tv, int hd)
{
    int i = blockIdx.x * blockDim.x + threadIdx.x;
    int total = Bv * H * Tv * hd;
    if (i >= total) return;
    int d = i % hd;
    int t = (i / hd) % Tv;
    int h = (i / (hd * Tv)) % H;
    int b = i / (hd * Tv * H);
    int C = H * hd;
    size_t base = ((size_t)b * Tv + t) * (size_t)(3 * C) + h * hd;
    int halfd = hd / 2;
    int j = d % halfd;
    float ang = (float)t * powf(10000.0f, (-2.0f * (float)j) / (float)hd);
    float cs = cosf(ang), sn = sinf(ang);

    float qv = (float)QKV[base + d];
    float qr = (d < halfd) ? -(float)QKV[base + d + halfd] : (float)QKV[base + d - halfd];
    Qo[i] = (_Float16)(qv * cs + qr * sn);
    float kv = (float)QKV[base + C + d];
    float kr = (d < halfd) ? -(float)QKV[base + C + d + halfd] : (float)QKV[base + C + d - halfd];
    Ko[i] = (_Float16)(kv * cs + kr * sn);
    Vo[i] = QKV[base + 2 * C + d];
}

// ======================= HBS mixer: Xout += hbs(Xin) =======================
__global__ __launch_bounds__(256)
void k_hbs(const float* __restrict__ Xin, float* __restrict__ Xout,
           const float* __restrict__ bw, const float* __restrict__ Wr,
           const float* __restrict__ Wc, const float* __restrict__ alpha,
           const float* __restrict__ bias, int D, int S)
{
    extern __shared__ float sm[];
    const int PM = S * S;
    float* xr  = sm;
    float* tmp = sm + PM;
    const int row = blockIdx.x;
    const float* x = Xin + (size_t)row * D;

    for (int j = threadIdx.x; j < PM; j += 256) xr[j] = (j < D) ? x[j] : 0.0f;
    __syncthreads();

    for (int i = threadIdx.x; i < S * S; i += 256) {
        int r = i / S, c = i % S;
        float acc = 0.0f;
        for (int k = 0; k < S; ++k) acc += xr[r * S + k] * Wc[k * S + c];
        tmp[i] = acc;
    }
    __syncthreads();

    float a = alpha[0];
    for (int j = threadIdx.x; j < D; j += 256) {
        int p = D - 1 - j;                     // row reversal of block output
        int g = p >> 4, c = p & 15;
        float yb = 0.0f;
        const float* xb  = xr + (g << 4);
        const float* bwp = bw + g * 256 + c;   // bw[g][b][c]
        for (int t = 0; t < 16; ++t) yb += xb[t] * bwp[t * 16];

        int r = j / S, cc = j % S;
        float mon = 0.0f;
        for (int k = 0; k < S; ++k) mon += tmp[r * S + k] * Wr[cc * S + k];

        Xout[(size_t)row * D + j] += yb + a * mon + bias[j];
    }
}

// ======================= token embedding gather ============================
__global__ void k_embed(const int* __restrict__ idx, const float* __restrict__ emb,
                        float* __restrict__ X, int NT, int D)
{
    int i = blockIdx.x * blockDim.x + threadIdx.x;
    if (i >= NT * D) return;
    int t = i / D, c = i % D;
    X[i] = emb[(size_t)idx[t] * D + c];
}

// ======================= compression einsum btkc,ock->bto ==================
__global__ void k_compress(const float* __restrict__ X, const float* __restrict__ Wc,
                           const float* __restrict__ bc, float* __restrict__ Mo,
                           int Bv, int TO, int LBSv, int DWv, int DMv)
{
    int i = blockIdx.x * blockDim.x + threadIdx.x;
    if (i >= Bv * TO * DMv) return;
    int o  = i % DMv;
    int to = (i / DMv) % TO;
    int b  = i / (DMv * TO);
    const float* xb = X + ((size_t)b * TO + to) * (size_t)LBSv * DWv;
    const float* w  = Wc + (size_t)o * DWv * LBSv;  // comp_w[o][c][k]
    float acc = bc[o];
    for (int k = 0; k < LBSv; ++k)
        for (int c = 0; c < DWv; ++c)
            acc += xb[(size_t)k * DWv + c] * w[(size_t)c * LBSv + k];
    Mo[i] = acc;
}

// ======================= VQ nearest-bucket =================================
__global__ void k_vq(const float* __restrict__ Mx, const float* __restrict__ emb,
                     float* __restrict__ Qo, int NR, int HV, int hdv, int NB)
{
    int i = blockIdx.x * blockDim.x + threadIdx.x;
    if (i >= NR * HV) return;
    int h = i % HV, row = i / HV;
    const float* f = Mx + (size_t)row * HV * hdv + h * hdv;
    int best = 0; float bd = 3.4e38f;
    for (int bkt = 0; bkt < NB; ++bkt) {
        const float* e = emb + ((size_t)h * NB + bkt) * hdv;
        float d2 = 0.0f;
        for (int d = 0; d < hdv; ++d) { float df = f[d] - e[d]; d2 += df * df; }
        if (d2 < bd) { bd = d2; best = bkt; }
    }
    const float* e = emb + ((size_t)h * NB + best) * hdv;
    for (int d = 0; d < hdv; ++d)
        Qo[(size_t)row * HV * hdv + h * hdv + d] = e[d];
}

// vq_loss = (1 + COMMIT) * mean((q - m)^2), single-block reduction.
__global__ __launch_bounds__(256)
void k_vqloss(const float* __restrict__ Qo, const float* __restrict__ Mx,
              float* __restrict__ out, int N)
{
    __shared__ float red[256];
    float s = 0.0f;
    for (int i = threadIdx.x; i < N; i += 256) { float d = Qo[i] - Mx[i]; s += d * d; }
    red[threadIdx.x] = s; __syncthreads();
    for (int o = 128; o > 0; o >>= 1) { if (threadIdx.x < o) red[threadIdx.x] += red[threadIdx.x + o]; __syncthreads(); }
    if (threadIdx.x == 0) out[0] = 1.5f * red[0] / (float)N;
}

// x[row][:] += ctx[row/64][:]
__global__ void k_addctx(float* __restrict__ X, const float* __restrict__ ctx,
                         int NT, int D, int LBSv)
{
    int i = blockIdx.x * blockDim.x + threadIdx.x;
    if (i >= NT * D) return;
    int c = i % D, t = i / D;
    X[i] += ctx[(size_t)(t / LBSv) * D + c];
}

// ===========================================================================
struct BlockP {
    const float *ln1_g, *ln1_b, *qkv_w, *qkv_b;
    const float *bw, *Wr, *Wc, *alpha, *bias;     // cproj (hbs)
    const float *ln2_g, *ln2_b, *fc_w, *fc_b, *proj_w, *proj_b;
};

extern "C" void kernel_launch(void* const* d_in, const int* in_sizes, int n_in,
                              void* d_out, int out_size, void* d_ws, size_t ws_size,
                              hipStream_t stream)
{
    (void)in_sizes; (void)n_in; (void)out_size; (void)ws_size;
    auto f = [&](int i) { return (const float*)d_in[i]; };

    const int*   idx     = (const int*)d_in[0];
    const float* tok_emb = f(1);
    int p = 2;
    BlockP wp[4], mp[4];
    for (int i = 0; i < 4; ++i) {
        wp[i] = { f(p+0), f(p+1), f(p+2), f(p+3), f(p+4), f(p+5), f(p+6), f(p+7),
                  f(p+8), f(p+9), f(p+10), f(p+11), f(p+12), f(p+13), f(p+14) };
        p += 15;
    }
    const float* comp_w = f(p++);
    const float* comp_b = f(p++);
    for (int i = 0; i < 4; ++i) {
        mp[i] = { f(p+0), f(p+1), f(p+2), f(p+3), f(p+4), f(p+5), f(p+6), f(p+7),
                  f(p+8), f(p+9), f(p+10), f(p+11), f(p+12), f(p+13), f(p+14) };
        p += 15;
    }
    const float* vq_emb = f(p++);
    const float* m2w_w  = f(p++);
    const float* m2w_b  = f(p++);
    const float* lnf_g  = f(p++);
    const float* lnf_b  = f(p++);
    const float* lm_w   = f(p++);

    float* out = (float*)d_out;
    float* ws  = (float*)d_ws;

    // ---- workspace layout ----
    const size_t NTOK = (size_t)BB * TT;                 // 4096
    float* x   = ws;                       // 1,048,576 f
    float* yb  = x  + 1048576;             // 1,048,576 f (attn out, word level)
    float* m   = yb + 1048576;             // 24,576 f (macro stream)
    float* my  = m  + 24576;               // 24,576 f (macro attn out)
    float* qv  = my + 24576;               // 24,576 f (VQ output)
    float* ctx = qv + 24576;               // 16,384 f
    float* b16 = ctx + 16384;              // f16 regions below (16B aligned)
    _Float16* xnh   = (_Float16*)b16;      // 1,048,576 h : LN outputs (GEMM A)
    _Float16* big16 = xnh   + 1048576;     // 4,194,304 h : qkv out / fc out
    _Float16* q16   = big16 + 4194304;     // 1,048,576 h
    _Float16* k16   = q16   + 1048576;     // 1,048,576 h
    _Float16* v16   = k16   + 1048576;     // 1,048,576 h
    _Float16* wh    = v16   + 1048576;     // 12,865,792 h : converted weights

    auto conv = [&](const float* src, _Float16* dst, size_t n) {
        int n4 = (int)(n >> 2);
        k_tof16<<<(n4 + 255) / 256, 256, 0, stream>>>(src, dst, n4);
    };

    auto run_block = [&](const BlockP& P, float* xv, float* ybv,
                         int Bv, int Tv, int Dv, int Hv, int Sv) {
        const int Mr = Bv * Tv;
        const int hd = Dv / Hv;
        k_ln<<<Mr, 256, 0, stream>>>(xv, P.ln1_g, P.ln1_b, xnh, Dv, 1);
        conv(P.qkv_w, wh, (size_t)3 * Dv * Dv);
        dim3 gq((3 * Dv) / 64, Mr / 64);
        k_gemm<<<gq, 256, 0, stream>>>(xnh, wh, P.qkv_b, nullptr, big16, Mr, 3 * Dv, Dv, 0, FLAG_HALF);
        int tot = Bv * Hv * Tv * hd;
        k_rope_split<<<(tot + 255) / 256, 256, 0, stream>>>(big16, q16, k16, v16, Bv, Hv, Tv, hd);
        dim3 ga(Tv / 16, Hv, Bv);
        k_attn<<<ga, 32, 0, stream>>>(q16, k16, v16, ybv, Bv, Hv, Tv, hd);
        k_hbs<<<Mr, 256, 2 * Sv * Sv * sizeof(float), stream>>>(ybv, xv, P.bw, P.Wr, P.Wc, P.alpha, P.bias, Dv, Sv);
        k_ln<<<Mr, 256, 0, stream>>>(xv, P.ln2_g, P.ln2_b, xnh, Dv, 1);
        conv(P.fc_w, wh, (size_t)4 * Dv * Dv);
        dim3 gf((4 * Dv) / 64, Mr / 64);
        k_gemm<<<gf, 256, 0, stream>>>(xnh, wh, P.fc_b, nullptr, big16, Mr, 4 * Dv, Dv, 1, FLAG_HALF);
        conv(P.proj_w, wh, (size_t)4 * Dv * Dv);
        dim3 gp(Dv / 64, Mr / 64);
        k_gemm<<<gp, 256, 0, stream>>>(big16, wh, P.proj_b, xv, xv, Mr, Dv, 4 * Dv, 0, 0);
    };

    // ---- word level ----
    k_embed<<<(int)((NTOK * DWD + 255) / 256), 256, 0, stream>>>(idx, tok_emb, x, (int)NTOK, DWD);
    for (int i = 0; i < 4; ++i)
        run_block(wp[i], x, yb, BB, TT, DWD, 4, 16);

    // ---- compress to macro stream ----
    {
        int tot = BB * (TT / LBSD) * DMD;
        k_compress<<<(tot + 255) / 256, 256, 0, stream>>>(x, comp_w, comp_b, m, BB, TT / LBSD, LBSD, DWD, DMD);
    }

    // ---- macro level (T=32, D=384, H=8, hd=48) ----
    for (int i = 0; i < 4; ++i)
        run_block(mp[i], m, my, BB, TT / LBSD, DMD, 8, 20);

    // ---- VQ + ctx projection + broadcast add ----
    k_vq<<<1, 256, 0, stream>>>(m, vq_emb, qv, BB * (TT / LBSD), 4, DMD / 4, 64);
    conv(qv, q16, 64 * DMD);                 // q16 free after macro attention
    conv(m2w_w, wh, (size_t)DWD * DMD);
    {
        dim3 gc(DWD / 64, 1);
        k_gemm<<<gc, 256, 0, stream>>>(q16, wh, m2w_b, nullptr, ctx, 64, DWD, DMD, 0, 0);
    }
    k_addctx<<<(int)((NTOK * DWD + 255) / 256), 256, 0, stream>>>(x, ctx, (int)NTOK, DWD, LBSD);

    // ---- final LN + LM head (f16 weights, NT stores for 823MB logits) ----
    k_ln<<<(int)NTOK, 256, 0, stream>>>(x, lnf_g, lnf_b, xnh, DWD, 1);
    conv(lm_w, wh, (size_t)VV * DWD);
    {
        dim3 gl((VV + 63) / 64, (int)NTOK / 64);
        k_gemm<<<gl, 256, 0, stream>>>(xnh, wh, nullptr, nullptr, out, (int)NTOK, VV, DWD, 0, FLAG_NT);
    }

    // ---- vq loss scalar ----
    k_vqloss<<<1, 256, 0, stream>>>(qv, m, out + (size_t)BB * TT * VV, 64 * DMD);
}
